// GNN_76175539962264
// MI455X (gfx1250) — matrix operand order repaired and checked by reading, not compiled, not run on hardware
//
#include <hip/hip_runtime.h>
#include <hip/hip_bf16.h>

typedef __attribute__((ext_vector_type(16))) _Float16 v16h;
typedef __attribute__((ext_vector_type(8)))  _Float16 v8h;
typedef __attribute__((ext_vector_type(8)))  float    v8f;

#define IN_DIM  128
#define HID     256
#define OUT_DIM 128
#define LN_EPS  1e-5f

// ---------------------------------------------------------------- utilities

__global__ void fill_zero_f32(float* __restrict__ p, int n) {
    int i = blockIdx.x * blockDim.x + threadIdx.x;
    if (i < n) p[i] = 0.0f;
}

// degree count: +1 per incoming edge (self loop added in finalize)
__global__ void deg_count(const int* __restrict__ dst, float* __restrict__ deg, int E) {
    int i = blockIdx.x * blockDim.x + threadIdx.x;
    if (i < E)
        __hip_atomic_fetch_add(&deg[dst[i]], 1.0f,
                               __ATOMIC_RELAXED, __HIP_MEMORY_SCOPE_AGENT);
}

// dinv = rsqrt(deg + 1)   (self loop guarantees deg >= 1, matches reference)
__global__ void finalize_dinv(float* __restrict__ deg, int N) {
    int i = blockIdx.x * blockDim.x + threadIdx.x;
    if (i < N) deg[i] = rsqrtf(deg[i] + 1.0f);
}

__global__ void f32_to_f16(const float* __restrict__ in, _Float16* __restrict__ out, long n) {
    long i = (long)blockIdx.x * blockDim.x + threadIdx.x;
    if (i < n) out[i] = (_Float16)in[i];
}

// Pack W [K x Nc] row-major f32 into per-lane WMMA B fragments (f16):
// tile (kk = k/32, nt = n/16): lane = (n%16) + 16*((k%32)/16), half j = k%16.
// B fragment for a wave = 512 contiguous halfs -> one aligned v16h load per lane.
__global__ void pack_w(const float* __restrict__ W, _Float16* __restrict__ Wp, int K, int Nc) {
    long i = (long)blockIdx.x * blockDim.x + threadIdx.x;
    long total = (long)K * Nc;
    if (i >= total) return;
    int k = (int)(i / Nc), n = (int)(i % Nc);
    int kk = k >> 5, kt = k & 31, nt = n >> 4, nl = n & 15;
    int lane = nl + ((kt >> 4) << 4);
    int j    = kt & 15;
    long didx = ((long)kk * (Nc >> 4) + nt) * 512 + lane * 16 + j;
    Wp[didx] = (_Float16)W[i];
}

// ---------------------------------------------------------------- WMMA GEMM
// C[M x Nc] = A[M x K] @ Wp (packed).  block = 256 threads = 8 waves.
// Each wave: 16 rows x 64 cols (4 WMMA tiles). M % 16 == 0, Nc % 64 == 0, K % 32 == 0.
__global__ void __launch_bounds__(256)
gemm_wmma_f16(const _Float16* __restrict__ A, const _Float16* __restrict__ Wp,
              float* __restrict__ C, int M, int K, int Nc) {
    const int wave = threadIdx.x >> 5;
    const int lane = threadIdx.x & 31;
    const int m0 = blockIdx.x * 128 + wave * 16;
    if (m0 >= M) return;                       // uniform per wave; full tiles otherwise
    const int n0 = blockIdx.y * 64;
    const int hi = lane >> 4;                  // 0 / 1 (half-wave)
    const int lo = lane & 15;

    v8f acc0 = {}, acc1 = {}, acc2 = {}, acc3 = {};

    const int row    = m0 + lo;                // A row for this lane
    const int kbase  = hi * 16;                // A: K split {0-15} / {16-31}
    const int ntiles = Nc >> 4;

    for (int kk = 0; kk < K; kk += 32) {
        const _Float16* ap = A + (long)row * K + kk + kbase;
        v8h alo = *(const v8h*)(ap);
        v8h ahi = *(const v8h*)(ap + 8);
        v16h a;
#pragma unroll
        for (int j = 0; j < 8; ++j) { a[j] = alo[j]; a[j + 8] = ahi[j]; }

        const _Float16* bp = Wp + ((long)(kk >> 5) * ntiles + (n0 >> 4)) * 512 + lane * 16;
        v16h b0 = *(const v16h*)(bp);
        v16h b1 = *(const v16h*)(bp + 512);
        v16h b2 = *(const v16h*)(bp + 1024);
        v16h b3 = *(const v16h*)(bp + 1536);

        acc0 = __builtin_amdgcn_wmma_f32_16x16x32_f16(false, a, false, b0, (short)0, acc0, false, false);
        acc1 = __builtin_amdgcn_wmma_f32_16x16x32_f16(false, a, false, b1, (short)0, acc1, false, false);
        acc2 = __builtin_amdgcn_wmma_f32_16x16x32_f16(false, a, false, b2, (short)0, acc2, false, false);
        acc3 = __builtin_amdgcn_wmma_f32_16x16x32_f16(false, a, false, b3, (short)0, acc3, false, false);
    }

    // C/D layout: VGPR r -> row m0 + r + 8*hi, col n = lo within each 16-wide tile
#pragma unroll
    for (int r = 0; r < 8; ++r) {
        long m = m0 + r + hi * 8;
        float* cr = C + m * (long)Nc + n0 + lo;
        cr[0]  = acc0[r];
        cr[16] = acc1[r];
        cr[32] = acc2[r];
        cr[48] = acc3[r];
    }
}

// ---------------------------------------------------------------- aggregation

// AGG = bias + self-loop message (T[v] * dinv[v]^2) — fuses bias add + self loops
__global__ void agg_init(const float* __restrict__ T, const float* __restrict__ dinv,
                         const float* __restrict__ bias, float* __restrict__ AGG,
                         int N, int D) {
    long i = (long)blockIdx.x * blockDim.x + threadIdx.x;
    long total = (long)N * D;
    if (i >= total) return;
    int v = (int)(i / D), c = (int)(i % D);
    float dv = dinv[v];
    AGG[i] = bias[c] + T[i] * dv * dv;
}

// one wave per edge: coalesced gather of T[src], hardware f32 atomics into AGG[dst]
__global__ void __launch_bounds__(256)
scatter_edges(const float* __restrict__ T, const int* __restrict__ src,
              const int* __restrict__ dst, const float* __restrict__ dinv,
              float* __restrict__ AGG, int E, int D) {
    int e = (int)(((long)blockIdx.x * blockDim.x + threadIdx.x) >> 5);
    int lane = threadIdx.x & 31;
    if (e >= E) return;
    int s = src[e], d = dst[e];
    float nrm = dinv[s] * dinv[d];
    const float* ts = T + (long)s * D;
    float* ad = AGG + (long)d * D;
    for (int c = lane; c < D; c += 32)
        __hip_atomic_fetch_add(ad + c, ts[c] * nrm,
                               __ATOMIC_RELAXED, __HIP_MEMORY_SCOPE_AGENT);
}

// ---------------------------------------------------------------- LN + ELU (+residual)
// one wave per node, D = HID = 256 (8 elems/lane), wave32 shfl reductions
__global__ void __launch_bounds__(256)
ln_elu_res(const float* __restrict__ AGG, const float* __restrict__ gamma,
           const float* __restrict__ beta, const float* __restrict__ res,
           float* __restrict__ out, int N) {
    int w = (int)(((long)blockIdx.x * blockDim.x + threadIdx.x) >> 5);
    int lane = threadIdx.x & 31;
    if (w >= N) return;
    const float* row = AGG + (long)w * HID;
    float v[8], s = 0.0f, ss = 0.0f;
#pragma unroll
    for (int j = 0; j < 8; ++j) {
        v[j] = row[lane + 32 * j];
        s += v[j];
        ss += v[j] * v[j];
    }
#pragma unroll
    for (int off = 16; off > 0; off >>= 1) {
        s  += __shfl_xor(s,  off, 32);
        ss += __shfl_xor(ss, off, 32);
    }
    float mu   = s  * (1.0f / HID);
    float var  = ss * (1.0f / HID) - mu * mu;
    float rstd = rsqrtf(var + LN_EPS);
    float* orow = out + (long)w * HID;
    const float* rrow = res ? res + (long)w * HID : nullptr;
#pragma unroll
    for (int j = 0; j < 8; ++j) {
        int c = lane + 32 * j;
        float y = (v[j] - mu) * rstd * gamma[c] + beta[c];
        y = (y > 0.0f) ? y : expm1f(y);   // ELU, alpha = 1
        if (rrow) y += rrow[c];
        orow[c] = y;
    }
}

// ---------------------------------------------------------------- orchestration

extern "C" void kernel_launch(void* const* d_in, const int* in_sizes, int n_in,
                              void* d_out, int out_size, void* d_ws, size_t ws_size,
                              hipStream_t stream) {
    const float* x  = (const float*)d_in[0];
    const int*   ei = (const int*)d_in[1];
    const float* Wm[4] = {(const float*)d_in[2], (const float*)d_in[4],
                          (const float*)d_in[6], (const float*)d_in[8]};
    const float* bv[4] = {(const float*)d_in[3], (const float*)d_in[5],
                          (const float*)d_in[7], (const float*)d_in[9]};
    const float* gv[3] = {(const float*)d_in[10], (const float*)d_in[12], (const float*)d_in[14]};
    const float* tv[3] = {(const float*)d_in[11], (const float*)d_in[13], (const float*)d_in[15]};

    const int N = in_sizes[0] / IN_DIM;   // 50000
    const int E = in_sizes[1] / 2;        // 800000
    const int* srcI = ei;
    const int* dstI = ei + E;

    // ---- carve workspace (256B aligned chunks) ----
    char* ws = (char*)d_ws;
    size_t off = 0;
    auto carve = [&](size_t bytes) -> void* {
        off = (off + 255) & ~(size_t)255;
        void* p = ws + off;
        off += bytes;
        return p;
    };
    float*    dinv = (float*)carve((size_t)N * sizeof(float));
    float*    T    = (float*)carve((size_t)N * HID * sizeof(float));
    float*    AGG  = (float*)carve((size_t)N * HID * sizeof(float));
    float*    Ha   = (float*)carve((size_t)N * HID * sizeof(float));
    float*    Hb   = (float*)carve((size_t)N * HID * sizeof(float));
    _Float16* Wp   = (_Float16*)carve((size_t)HID * HID * sizeof(_Float16));
    _Float16* Hh   = (_Float16*)AGG;   // alias: Hh dead before AGG is first written each layer

    const int TB = 256;

    // ---- symmetric normalization ----
    fill_zero_f32<<<(N + TB - 1) / TB, TB, 0, stream>>>(dinv, N);
    deg_count    <<<(E + TB - 1) / TB, TB, 0, stream>>>(dstI, dinv, E);
    finalize_dinv<<<(N + TB - 1) / TB, TB, 0, stream>>>(dinv, N);

    // ---- 4 conv layers ----
    const int Ks[4]  = {IN_DIM, HID, HID, HID};
    const int Ncs[4] = {HID, HID, HID, OUT_DIM};
    const float* Hcur = x;
    float* Hbufs[2] = {Ha, Hb};

    for (int l = 0; l < 4; ++l) {
        const int K = Ks[l], Nc = Ncs[l];

        long cntA = (long)N * K;
        f32_to_f16<<<(unsigned)((cntA + TB - 1) / TB), TB, 0, stream>>>(Hcur, Hh, cntA);

        long cntW = (long)K * Nc;
        pack_w<<<(unsigned)((cntW + TB - 1) / TB), TB, 0, stream>>>(Wm[l], Wp, K, Nc);

        dim3 gg((N + 127) / 128, Nc / 64);
        gemm_wmma_f16<<<gg, TB, 0, stream>>>(Hh, Wp, T, N, K, Nc);

        float* aggdst = (l == 3) ? (float*)d_out : AGG;

        long cntO = (long)N * Nc;
        agg_init<<<(unsigned)((cntO + TB - 1) / TB), TB, 0, stream>>>(T, dinv, bv[l], aggdst, N, Nc);

        long sThreads = (long)E * 32;
        scatter_edges<<<(unsigned)((sThreads + TB - 1) / TB), TB, 0, stream>>>(
            T, srcI, dstI, dinv, aggdst, E, Nc);

        if (l < 3) {
            const float* res = (l == 0) ? nullptr : Hcur;
            float* Hnext = Hbufs[l & 1];      // l0->Ha, l1->Hb, l2->Ha
            long lThreads = (long)N * 32;
            ln_elu_res<<<(unsigned)((lThreads + TB - 1) / TB), TB, 0, stream>>>(
                aggdst, gv[l], tv[l], res, Hnext, N);
            Hcur = Hnext;
        }
    }
}
